// DeltaNet_22488448762160
// MI455X (gfx1250) — compile-verified, matrix operand order checked by reading, not run on hardware
//
#include <hip/hip_runtime.h>
#include <hip/hip_bf16.h>

// ---------------------------------------------------------------------------
// DeltaNet-style gated linear attention for MI455X (gfx1250).
//   q/k/v/out projections  -> bf16 WMMA 16x16x32 GEMMs (f32 accum, reg-direct)
//   beta/mix gates         -> VALU dot products + sigmoid/softmax
//   RoPE                   -> elementwise f32
//   recurrence over T      -> CHUNKED formulation (L=32): all heavy math as
//                             bf16 WMMA against an LDS-resident bf16 state,
//                             q/k/v staged through LDS with async-to-LDS
//                             loads (ASYNCcnt) when the toolchain has them.
// ---------------------------------------------------------------------------

#define Hh 8
#define Kk 2
#define Dd 128
#define Tt 2048
#define Bb 4
#define Cc 1024
#define BT (Bb * Tt)
#define CL 32                      // chunk length
#define NCH (Tt / CL)              // 64 chunks

typedef __attribute__((ext_vector_type(16))) __bf16       v16bf;
typedef __attribute__((ext_vector_type(8)))  float        v8f;
typedef __attribute__((ext_vector_type(8)))  unsigned int v8u;
typedef __attribute__((ext_vector_type(4)))  int          v4i;

#if __has_builtin(__builtin_amdgcn_global_load_async_to_lds_b128) && \
    __has_builtin(__builtin_amdgcn_s_wait_asynccnt)
#define USE_ASYNC_LDS 1
#else
#define USE_ASYNC_LDS 0
#endif

#define AS1 __attribute__((address_space(1)))
#define AS3 __attribute__((address_space(3)))

__device__ __forceinline__ unsigned short f2bf(float f) {
    unsigned int u = __builtin_bit_cast(unsigned int, f);
    unsigned int r = u + 0x7FFFu + ((u >> 16) & 1u);   // round-to-nearest-even
    return (unsigned short)(r >> 16);
}
__device__ __forceinline__ float bf2f(unsigned short b) {
    unsigned int u = ((unsigned int)b) << 16;
    return __builtin_bit_cast(float, u);
}

// 16-element bf16 fragment from two contiguous 16B runs (k..k+7, k+16..k+23).
__device__ __forceinline__ v16bf load_frag(const unsigned short* p) {
    uint4 r1 = *(const uint4*)(p);
    uint4 r2 = *(const uint4*)(p + 16);
    v8u u;
    u[0] = r1.x; u[1] = r1.y; u[2] = r1.z; u[3] = r1.w;
    u[4] = r2.x; u[5] = r2.y; u[6] = r2.z; u[7] = r2.w;
    return __builtin_bit_cast(v16bf, u);
}

// Column gather: fragment for A-operand row `dcol` of K'^T, i.e. K'[i][dcol],
// from row-major K' [32][128] in LDS. kb = 8*laneHi.
__device__ __forceinline__ v16bf gather_colT(const unsigned short* Kpb, int dcol, int kb) {
    v8u u;
#pragma unroll
    for (int w = 0; w < 8; ++w) {
        int i0 = ((w < 4) ? 2 * w : 2 * w + 8) + kb;   // pairs (0,1)..(6,7),(16,17)..(22,23)
        unsigned lo = Kpb[i0 * Dd + dcol];
        unsigned hi = Kpb[(i0 + 1) * Dd + dcol];
        u[w] = lo | (hi << 16);
    }
    return __builtin_bit_cast(v16bf, u);
}

#define WMMA_BF16(a, b, c) \
    __builtin_amdgcn_wmma_f32_16x16x32_bf16(false, (a), false, (b), (short)0, (c), false, false)

// --------------------------- elementwise helpers ---------------------------

__global__ void cast_f32_to_bf16(const float* __restrict__ src,
                                 unsigned short* __restrict__ dst, int n) {
    int i = blockIdx.x * blockDim.x + threadIdx.x;
    if (i < n) dst[i] = f2bf(src[i]);
}

__global__ void wtrans_cast(const float* __restrict__ W,
                            unsigned short* __restrict__ Wt, int Kd, int Nd) {
    int i = blockIdx.x * blockDim.x + threadIdx.x;
    if (i >= Kd * Nd) return;
    int n = i / Kd, kx = i % Kd;
    Wt[(size_t)n * Kd + kx] = f2bf(W[(size_t)kx * Nd + n]);
}

__global__ void zero_f32(float* p, int n) {
    int i = blockIdx.x * blockDim.x + threadIdx.x;
    if (i < n) p[i] = 0.0f;
}

// --------------------------- WMMA bf16 GEMM --------------------------------
// C[M,N] f32 = A[M,K]bf16 * Bt[N,K]bf16^T + bias. Wave: 32x32 (2x2 WMMAs).

__global__ void __launch_bounds__(256)
gemm_bf16_wmma(const unsigned short* __restrict__ A,
               const unsigned short* __restrict__ Bt,
               const float* __restrict__ bias,
               float* __restrict__ C, int M, int N, int Kd) {
    const int wave   = threadIdx.x >> 5;
    const int lane   = threadIdx.x & 31;
    const int laneHi = lane >> 4;
    const int ln     = lane & 15;
    const int wm = wave >> 2;
    const int wn = wave & 3;
    const int M0 = blockIdx.y * 64 + wm * 32;
    const int N0 = blockIdx.x * 128 + wn * 32;
    const int kb = laneHi ? 8 : 0;

    v8f acc00 = {}, acc01 = {}, acc10 = {}, acc11 = {};
    const unsigned short* arow0 = A + (size_t)(M0 + ln) * Kd + kb;
    const unsigned short* arow1 = A + (size_t)(M0 + 16 + ln) * Kd + kb;
    const unsigned short* brow0 = Bt + (size_t)(N0 + ln) * Kd + kb;
    const unsigned short* brow1 = Bt + (size_t)(N0 + 16 + ln) * Kd + kb;

    for (int k0 = 0; k0 < Kd; k0 += 32) {
        v16bf a0 = load_frag(arow0 + k0);
        v16bf a1 = load_frag(arow1 + k0);
        v16bf b0 = load_frag(brow0 + k0);
        v16bf b1 = load_frag(brow1 + k0);
        acc00 = WMMA_BF16(a0, b0, acc00);
        acc01 = WMMA_BF16(a0, b1, acc01);
        acc10 = WMMA_BF16(a1, b0, acc10);
        acc11 = WMMA_BF16(a1, b1, acc11);
    }
    const int mbase = M0 + laneHi * 8;
    int n0 = N0 + ln, n1 = N0 + 16 + ln;
    float bv0 = bias ? bias[n0] : 0.0f;
    float bv1 = bias ? bias[n1] : 0.0f;
#pragma unroll
    for (int r = 0; r < 8; ++r) {
        C[(size_t)(mbase + r) * N + n0]      = acc00[r] + bv0;
        C[(size_t)(mbase + r) * N + n1]      = acc01[r] + bv1;
        C[(size_t)(mbase + 16 + r) * N + n0] = acc10[r] + bv0;
        C[(size_t)(mbase + 16 + r) * N + n1] = acc11[r] + bv1;
    }
}

// --------------------------- beta / mix gates ------------------------------

__global__ void beta_mix_kernel(const float* __restrict__ x,
                                const float* __restrict__ Wb, const float* __restrict__ bb,
                                const float* __restrict__ Wm, const float* __restrict__ bm,
                                float* __restrict__ beta, float* __restrict__ mixo,
                                int rows) {
    int t = blockIdx.x * blockDim.x + threadIdx.x;
    if (t >= rows * Hh) return;
    int row = t >> 3, h = t & 7;
    float db0 = bb[h * 2], db1 = bb[h * 2 + 1];
    float dm0 = bm[h * 2], dm1 = bm[h * 2 + 1];
    const float* xr = x + (size_t)row * Cc;
    for (int c = 0; c < Cc; ++c) {
        float xv = xr[c];
        db0 = fmaf(xv, Wb[c * 16 + h * 2], db0);
        db1 = fmaf(xv, Wb[c * 16 + h * 2 + 1], db1);
        dm0 = fmaf(xv, Wm[c * 16 + h * 2], dm0);
        dm1 = fmaf(xv, Wm[c * 16 + h * 2 + 1], dm1);
    }
    float b0 = 1.0f / (1.0f + __expf(-db0));
    float b1 = 1.0f / (1.0f + __expf(-db1));
    float mx = fmaxf(dm0, dm1);
    float e0 = __expf(dm0 - mx), e1 = __expf(dm1 - mx);
    float inv = 1.0f / (e0 + e1);
    size_t o = (size_t)row * 16 + h * 2;
    beta[o] = b0;       beta[o + 1] = b1;
    mixo[o] = e0 * inv; mixo[o + 1] = e1 * inv;
}

// --------------------------- RoPE ------------------------------------------

__global__ void rope_kernel(float* __restrict__ qk) {
    int idx = blockIdx.x * blockDim.x + threadIdx.x;
    if (idx >= BT * (Cc / 2)) return;
    int row = idx / (Cc / 2);
    int p   = idx % (Cc / 2);
    int j   = p & 63;
    int t   = row % Tt;
    float freq = (float)t * __powf(10000.0f, -((float)(2 * j)) / 128.0f);
    float s, c;
    __sincosf(freq, &s, &c);
    float* ptr = qk + (size_t)row * Cc + p * 2;
    float xr = ptr[0], xi = ptr[1];
    ptr[0] = xr * c - xi * s;
    ptr[1] = xr * s + xi * c;
}

// --------------------------- chunked recurrent scan ------------------------
// grid = B*H*K = 64 blocks, 256 threads (8 waves).
// State S[d][e] kept as bf16 in LDS (St[e][d], 32 KB), updated per 32-chunk:
//   A  = Q K'^T            (K'_i = (P_L/P_i) k_i; masked rows: beta=1, k=0)
//   Am = causal_mask(A) * P_t/P_L          -> bf16 tile
//   Yi = Am V              (intra-chunk)
//   Yr = Q S0              (inter-chunk; scaled per-row by P_t)
//   S  = P_L S0 + K'^T V
// q/k/v staged via async global->LDS b128 loads where available.

__global__ void __launch_bounds__(256)
scan_chunk_kernel(const float* __restrict__ q, const float* __restrict__ k,
                  const float* __restrict__ v, const float* __restrict__ beta,
                  const float* __restrict__ mixp, const int* __restrict__ mask,
                  float* __restrict__ y) {
    __shared__ unsigned short St[Dd * Dd];    // 32 KB: S^T, [e][d] bf16
    __shared__ unsigned short Qb[CL * Dd];    // 8 KB  [t][d]
    __shared__ unsigned short Kpb[CL * Dd];   // 8 KB  [i][d] (decay-scaled)
    __shared__ unsigned short Vt[Dd * CL];    // 8 KB  [e][i]
    __shared__ float stageA[1024];            // 4 KB staging (overlaid with Ab)
    __shared__ float beL[CL], PL[CL], mxL[CL];
    __shared__ int   mskL[CL];
    unsigned short* Ab = (unsigned short*)stageA;   // [t][i] 2 KB, dead at staging time

    const int bid = blockIdx.x;               // ((b*H)+h)*K + kk
    const int kk = bid & 1;
    const int h  = (bid >> 1) & 7;
    const int b  = bid >> 4;
    const int tid  = threadIdx.x;
    const int wave = tid >> 5;
    const int lane = tid & 31;
    const int hi   = lane >> 4;
    const int ln   = lane & 15;
    const int kb   = hi * 8;

    for (int i = tid; i < Dd * Dd; i += 256) St[i] = 0;
    __syncthreads();

    for (int ch = 0; ch < NCH; ++ch) {
        const int rowbase = b * Tt + ch * CL;

        // ---- decay state ------------------------------------------------
        if (tid < CL) {
            int row = rowbase + tid;
            int mk  = mask[row];
            mskL[tid] = mk;
            beL[tid]  = (mk > 0) ? beta[(size_t)row * 16 + h * 2 + kk] : 1.0f;
            mxL[tid]  = mixp[(size_t)row * 16 + h * 2 + kk];
        }
        __syncthreads();
        if (tid < CL) {
            float p = 1.0f;
            for (int i = 0; i <= tid; ++i) p *= beL[i];
            PL[tid] = p;
        }
        __syncthreads();
        const float pl     = PL[CL - 1];
        const float inv_pl = 1.0f / pl;

        // ---- stage q/k/v chunk: 4 rounds of 8 rows, 16B per thread ------
        float* slot = stageA + tid * 4;
#pragma unroll
        for (int rnd = 0; rnd < 4; ++rnd) {
            const int rloc = rnd * 8 + (tid >> 5);
            const int col  = (tid & 31) * 4;
            const size_t gbase = (size_t)(rowbase + rloc) * Cc + h * Dd + col;
#pragma unroll
            for (int which = 0; which < 3; ++which) {
                const float* src = (which == 0) ? q : (which == 1) ? k : v;
#if USE_ASYNC_LDS
                __builtin_amdgcn_global_load_async_to_lds_b128(
                    (AS1 v4i*)(src + gbase), (AS3 v4i*)slot, 0, 0);
                __builtin_amdgcn_s_wait_asynccnt(0);
                asm volatile("" ::: "memory");
                float4 d4 = *(float4*)slot;
#else
                float4 d4 = *(const float4*)(src + gbase);
#endif
                float vals[4] = {d4.x, d4.y, d4.z, d4.w};
                if (which == 0) {
#pragma unroll
                    for (int j = 0; j < 4; ++j)
                        Qb[rloc * Dd + col + j] = f2bf(vals[j]);
                } else if (which == 1) {
                    float sc = (mskL[rloc] > 0) ? (pl / PL[rloc]) : 0.0f;
#pragma unroll
                    for (int j = 0; j < 4; ++j)
                        Kpb[rloc * Dd + col + j] = f2bf(vals[j] * sc);
                } else {
#pragma unroll
                    for (int j = 0; j < 4; ++j)
                        Vt[(col + j) * CL + rloc] = f2bf(vals[j]);
                }
            }
        }
        __syncthreads();

        // ---- GEMM1: A = Q K'^T  (waves 0-3, one 16x16 tile each) --------
        if (wave < 4) {
            const int tm = wave >> 1, tn = wave & 1;
            v8f a = {};
            const unsigned short* ar = Qb  + (tm * 16 + ln) * Dd + kb;
            const unsigned short* br = Kpb + (tn * 16 + ln) * Dd + kb;
#pragma unroll
            for (int k0 = 0; k0 < Dd; k0 += 32)
                a = WMMA_BF16(load_frag(ar + k0), load_frag(br + k0), a);
            const int iloc = tn * 16 + ln;
#pragma unroll
            for (int r = 0; r < 8; ++r) {
                int t = tm * 16 + hi * 8 + r;
                float val = (iloc <= t) ? a[r] * PL[t] * inv_pl : 0.0f;
                Ab[t * CL + iloc] = f2bf(val);
            }
        }
        __syncthreads();

        // ---- GEMM2: Yi = Am V   &   GEMM4: Yr = Q S0 (tiles (tm, wave)) --
        v8f yi0 = {}, yi1 = {}, yr0 = {}, yr1 = {};
        {
            const unsigned short* vr = Vt + (wave * 16 + ln) * CL + kb;
            v16bf bvf = load_frag(vr);
            yi0 = WMMA_BF16(load_frag(Ab + (0 * 16 + ln) * CL + kb), bvf, yi0);
            yi1 = WMMA_BF16(load_frag(Ab + (1 * 16 + ln) * CL + kb), bvf, yi1);
            const unsigned short* q0 = Qb + (0 * 16 + ln) * Dd + kb;
            const unsigned short* q1 = Qb + (1 * 16 + ln) * Dd + kb;
            const unsigned short* sr = St + (wave * 16 + ln) * Dd + kb;
#pragma unroll
            for (int k0 = 0; k0 < Dd; k0 += 32) {
                v16bf sf = load_frag(sr + k0);
                yr0 = WMMA_BF16(load_frag(q0 + k0), sf, yr0);
                yr1 = WMMA_BF16(load_frag(q1 + k0), sf, yr1);
            }
        }
        __syncthreads();   // all S0 reads done before update writes

        // ---- GEMM3: S = pl*S0 + K'^T V  (wave owns d-rows [16w,16w+16)) --
        {
            const int dbase = wave * 16;
            v16bf af = gather_colT(Kpb, dbase + ln, kb);
#pragma unroll
            for (int et = 0; et < 8; ++et) {
                v16bf bvf = load_frag(Vt + (et * 16 + ln) * CL + kb);
                v8f cc;
#pragma unroll
                for (int r = 0; r < 8; ++r) {
                    int d = dbase + hi * 8 + r;
                    int e = et * 16 + ln;
                    cc[r] = pl * bf2f(St[e * Dd + d]);
                }
                cc = WMMA_BF16(af, bvf, cc);
#pragma unroll
                for (int r = 0; r < 8; ++r) {
                    int d = dbase + hi * 8 + r;
                    int e = et * 16 + ln;
                    St[e * Dd + d] = f2bf(cc[r]);
                }
            }
        }
        __syncthreads();

        // ---- combine + emit y ------------------------------------------
        {
            const int e = wave * 16 + ln;
#pragma unroll
            for (int tm = 0; tm < 2; ++tm) {
                const v8f& yi = tm ? yi1 : yi0;
                const v8f& yr = tm ? yr1 : yr0;
#pragma unroll
                for (int r = 0; r < 8; ++r) {
                    int t = tm * 16 + hi * 8 + r;
                    if (mskL[t] > 0) {
                        float val = mxL[t] * (yi[r] + PL[t] * yr[r]);
                        atomicAdd(&y[(size_t)(rowbase + t) * Cc + h * Dd + e], val);
                    }
                }
            }
        }
        // next-chunk prefetch (global_prefetch_b8)
        if (ch + 1 < NCH) {
            size_t nb = (size_t)(rowbase + CL + (tid >> 3)) * Cc + h * Dd + (tid & 7) * 16;
            __builtin_prefetch(q + nb, 0, 1);
            __builtin_prefetch(k + nb, 0, 1);
            __builtin_prefetch(v + nb, 0, 1);
        }
        __syncthreads();   // tiles free for next chunk's staging
    }
}

// --------------------------- launcher --------------------------------------

extern "C" void kernel_launch(void* const* d_in, const int* in_sizes, int n_in,
                              void* d_out, int out_size, void* d_ws, size_t ws_size,
                              hipStream_t stream) {
    const float* x    = (const float*)d_in[0];
    const int*   mask = (const int*)d_in[1];
    const float* Wq   = (const float*)d_in[2];
    const float* Wk   = (const float*)d_in[3];
    const float* Wv   = (const float*)d_in[4];
    const float* Wb   = (const float*)d_in[5];
    const float* bb   = (const float*)d_in[6];
    const float* Wm   = (const float*)d_in[7];
    const float* bm   = (const float*)d_in[8];
    const float* Wo   = (const float*)d_in[9];
    const float* bo   = (const float*)d_in[10];
    float* out = (float*)d_out;

    char* w = (char*)d_ws;
    size_t off = 0;
    auto alloc = [&](size_t bytes) { char* p = w + off; off += (bytes + 255) & ~size_t(255); return p; };
    unsigned short* xbf = (unsigned short*)alloc((size_t)BT * Cc * 2);
    unsigned short* WqT = (unsigned short*)alloc((size_t)Cc * Cc * 2);
    unsigned short* WkT = (unsigned short*)alloc((size_t)Cc * Cc * 2);
    unsigned short* WvT = (unsigned short*)alloc((size_t)Cc * Cc * 2);
    unsigned short* WoT = (unsigned short*)alloc((size_t)Cc * Cc * 2);
    float* qf   = (float*)alloc((size_t)BT * Cc * 4);
    float* kf   = (float*)alloc((size_t)BT * Cc * 4);
    float* vf   = (float*)alloc((size_t)BT * Cc * 4);
    float* beta = (float*)alloc((size_t)BT * Hh * Kk * 4);
    float* mixp = (float*)alloc((size_t)BT * Hh * Kk * 4);
    float* yf   = (float*)alloc((size_t)BT * Cc * 4);
    unsigned short* ybf = (unsigned short*)alloc((size_t)BT * Cc * 2);
    (void)ws_size;

    const int nelem = BT * Cc;
    cast_f32_to_bf16<<<nelem / 256, 256, 0, stream>>>(x, xbf, nelem);
    wtrans_cast<<<(Cc * Cc) / 256, 256, 0, stream>>>(Wq, WqT, Cc, Cc);
    wtrans_cast<<<(Cc * Cc) / 256, 256, 0, stream>>>(Wk, WkT, Cc, Cc);
    wtrans_cast<<<(Cc * Cc) / 256, 256, 0, stream>>>(Wv, WvT, Cc, Cc);
    wtrans_cast<<<(Cc * Cc) / 256, 256, 0, stream>>>(Wo, WoT, Cc, Cc);

    dim3 ggrid(Cc / 128, BT / 64);
    gemm_bf16_wmma<<<ggrid, 256, 0, stream>>>(xbf, WqT, nullptr, qf, BT, Cc, Cc);
    gemm_bf16_wmma<<<ggrid, 256, 0, stream>>>(xbf, WkT, nullptr, kf, BT, Cc, Cc);
    gemm_bf16_wmma<<<ggrid, 256, 0, stream>>>(xbf, WvT, nullptr, vf, BT, Cc, Cc);

    beta_mix_kernel<<<(BT * Hh) / 256, 256, 0, stream>>>(x, Wb, bb, Wm, bm, beta, mixp, BT);

    const int npairs = BT * (Cc / 2);
    rope_kernel<<<npairs / 256, 256, 0, stream>>>(qf);
    rope_kernel<<<npairs / 256, 256, 0, stream>>>(kf);

    zero_f32<<<nelem / 256, 256, 0, stream>>>(yf, nelem);
    scan_chunk_kernel<<<Bb * Hh * Kk, 256, 0, stream>>>(qf, kf, vf, beta, mixp, mask, yf);

    cast_f32_to_bf16<<<nelem / 256, 256, 0, stream>>>(yf, ybf, nelem);
    gemm_bf16_wmma<<<ggrid, 256, 0, stream>>>(ybf, WoT, bo, out, BT, Cc, Cc);
}
